// PVConvReLU_4148938408186
// MI455X (gfx1250) — matrix operand order repaired and checked by reading, not compiled, not run on hardware
//
#include <hip/hip_runtime.h>
#include <hip/hip_bf16.h>

typedef __attribute__((ext_vector_type(16))) _Float16 v16h;
typedef __attribute__((ext_vector_type(8)))  _Float16 v8h;
typedef __attribute__((ext_vector_type(8)))  float    v8f;

#define BN_  8
#define CIN_ 64
#define COUT_ 64
#define NPTS 4096
#define RV 32
#define R3 32768
#define LEAK 0.2f
#define EPS_ 1e-5f

// param-block float offsets (within ws param region)
#define O_BN1SC 0
#define O_BN1BS 64
#define O_BN2SC 128
#define O_BN2BS 192
#define O_MEAN  256   // 24 floats
#define O_SCALE 280   // 8 floats
#define O_GNMU  288   // 64 floats
#define O_GNINV 352   // 64 floats

// ---------------- prep: fold BN into per-channel scale/bias ----------------
__global__ void prep_params(const float* __restrict__ c1b, const float* __restrict__ g1,
                            const float* __restrict__ b1, const float* __restrict__ m1,
                            const float* __restrict__ v1,
                            const float* __restrict__ c2b, const float* __restrict__ g2,
                            const float* __restrict__ b2, const float* __restrict__ m2,
                            const float* __restrict__ v2, float* par) {
  int c = threadIdx.x;
  if (c < 64) {
    float s1 = g1[c] * rsqrtf(v1[c] + EPS_);
    par[O_BN1SC + c] = s1;
    par[O_BN1BS + c] = (c1b[c] - m1[c]) * s1 + b1[c];
    float s2 = g2[c] * rsqrtf(v2[c] + EPS_);
    par[O_BN2SC + c] = s2;
    par[O_BN2BS + c] = (c2b[c] - m2[c]) * s2 + b2[c];
  }
}

// ---------------- pack weights into WMMA A-fragment lane layout ----------------
// conv: k = kk*64 + ci (kernel-position-major). step s = kk*2 + (c0/32).
// A 16x32 f16 packing used here (contiguous k, same order as the B loads): lane L, half h -> M = L%16, K = h + (L>=16)*16
__global__ void pack_weights(const float* __restrict__ w1, const float* __restrict__ w2,
                             const float* __restrict__ pfw,
                             _Float16* __restrict__ pw1, _Float16* __restrict__ pw2,
                             _Float16* __restrict__ ppf) {
  int t = blockIdx.x * blockDim.x + threadIdx.x;
  const int NC = 4 * 54 * 32 * 16;  // 110592
  if (t < 2 * NC) {
    int u = (t < NC) ? t : t - NC;
    int h = u & 15, L = (u >> 4) & 31, s = (u >> 9) % 54, mt = (u >> 9) / 54;
    int kk = s >> 1, c0 = (s & 1) * 32;
    int K = h + ((L >= 16) ? 16 : 0);
    int ci = c0 + K, co = mt * 16 + (L & 15);
    const float* src = (t < NC) ? w1 : w2;
    _Float16* dst = (t < NC) ? pw1 : pw2;
    dst[u] = (_Float16)src[(co * 64 + ci) * 27 + kk];
  } else if (t < 2 * NC + 4096) {
    int u = t - 2 * NC;
    int h = u & 15, L = (u >> 4) & 31, s = (u >> 9) & 1, mt = (u >> 9) >> 1;
    int K = h + ((L >= 16) ? 16 : 0);
    int ci = s * 32 + K, co = mt * 16 + (L & 15);
    ppf[u] = (_Float16)pfw[co * 64 + ci];
  }
}

// ---------------- per-batch coordinate mean / scale ----------------
__global__ void coord_stats(const float* __restrict__ coords, float* par) {
  __shared__ float red[256];
  __shared__ float m3[3];
  int b = blockIdx.x, tid = threadIdx.x;
  for (int a = 0; a < 3; ++a) {
    float s = 0.f;
    for (int n = tid; n < NPTS; n += 256) s += coords[(b * 3 + a) * NPTS + n];
    red[tid] = s; __syncthreads();
    for (int off = 128; off > 0; off >>= 1) {
      if (tid < off) red[tid] += red[tid + off];
      __syncthreads();
    }
    if (tid == 0) m3[a] = red[0] / (float)NPTS;
    __syncthreads();
  }
  float mx = 0.f;
  for (int n = tid; n < NPTS; n += 256) {
    float dx = coords[(b * 3 + 0) * NPTS + n] - m3[0];
    float dy = coords[(b * 3 + 1) * NPTS + n] - m3[1];
    float dz = coords[(b * 3 + 2) * NPTS + n] - m3[2];
    mx = fmaxf(mx, sqrtf(dx * dx + dy * dy + dz * dz));
  }
  red[tid] = mx; __syncthreads();
  for (int off = 128; off > 0; off >>= 1) {
    if (tid < off) red[tid] = fmaxf(red[tid], red[tid + off]);
    __syncthreads();
  }
  if (tid == 0) {
    par[O_SCALE + b] = red[0] * 2.f;
    par[O_MEAN + b * 3 + 0] = m3[0];
    par[O_MEAN + b * 3 + 1] = m3[1];
    par[O_MEAN + b * 3 + 2] = m3[2];
  }
}

// ---------------- voxelize: scatter-add + norm coords ----------------
__global__ void voxelize(const float* __restrict__ coords, const float* __restrict__ features,
                         const float* __restrict__ par, float* __restrict__ norm_c,
                         float* __restrict__ cnt, float* __restrict__ sums) {
  int t = blockIdx.x * blockDim.x + threadIdx.x;   // B*NPTS
  int b = t >> 12, n = t & (NPTS - 1);
  float inv_s = 1.f / par[O_SCALE + b];
  int vox[3];
  for (int a = 0; a < 3; ++a) {
    float c = coords[(b * 3 + a) * NPTS + n];
    float nc = (c - par[O_MEAN + b * 3 + a]) * inv_s + 0.5f;
    nc = fminf(fmaxf(nc * (float)RV, 0.f), (float)(RV - 1));
    norm_c[(b * 3 + a) * NPTS + n] = nc;
    vox[a] = (int)roundf(nc);
  }
  int idx = vox[0] * (RV * RV) + vox[1] * RV + vox[2];
  atomicAdd(cnt + b * R3 + idx, 1.f);
  float* sp = sums + ((size_t)(b * R3 + idx)) * 64;
  for (int ci = 0; ci < 64; ++ci)
    atomicAdd(sp + ci, features[(b * 64 + ci) * NPTS + n]);
}

// ---------------- mean-divide + f16 convert, channels-last grid ----------------
__global__ void finalize_grid(const float* __restrict__ sums, const float* __restrict__ cnt,
                              _Float16* __restrict__ g1) {
  int t = blockIdx.x * blockDim.x + threadIdx.x;   // B*R3
  float inv = 1.f / fmaxf(cnt[t], 1.f);
  size_t base = (size_t)t * 64;
  for (int ci = 0; ci < 64; ++ci)
    g1[base + ci] = (_Float16)(sums[base + ci] * inv);
}

// ---------------- implicit-GEMM 3x3x3 conv via WMMA f16->f32 ----------------
// one wave computes ALL FOUR 16-co tiles for one 16-z column tile:
// per K-step: 1 B-fragment load feeds 4 independent WMMA accumulation chains.
__global__ void conv3d_wmma(const _Float16* __restrict__ gin, const _Float16* __restrict__ pw,
                            const float* __restrict__ par, int sc_off, int bs_off,
                            _Float16* __restrict__ gout_h, float* __restrict__ gout_f) {
  int tid = threadIdx.x;
  int lane = tid & 31;
  int w = blockIdx.x * 8 + (tid >> 5);   // 16384 waves total
  int zt = w & 1; int yy = (w >> 1) & 31;
  int xx = (w >> 6) & 31; int b = w >> 11;
  int lm = lane & 15, hi = lane >> 4;
  int zl = zt * 16 + lm;              // this lane's z column (N index)
  int ci0 = hi * 16;                  // B-frag: K = h + hi*16

  v8f acc0 = {}, acc1 = {}, acc2 = {}, acc3 = {};
  const _Float16* pwl = pw + lane * 16;

  for (int dx = 0; dx < 3; ++dx) {
    int gx = xx + dx - 1;
    bool xok = ((unsigned)gx < 32u);
    for (int dy = 0; dy < 3; ++dy) {
      int gy = yy + dy - 1;
      bool yok = xok && ((unsigned)gy < 32u);
      for (int dz = 0; dz < 3; ++dz) {
        int gz = zl + dz - 1;
        bool ok = yok && ((unsigned)gz < 32u);
        int kk = (dx * 3 + dy) * 3 + dz;
        long long vbase = ((((long long)b * 32 + gx) * 32 + gy) * 32 + gz) * 64;
        #pragma unroll
        for (int s2 = 0; s2 < 2; ++s2) {
          v16h bfrag = {};
          if (ok) bfrag = *(const v16h*)(gin + vbase + s2 * 32 + ci0);
          const _Float16* ap = pwl + (size_t)(kk * 2 + s2) * 512;
          v16h a0 = *(const v16h*)(ap);
          v16h a1 = *(const v16h*)(ap + 54 * 512);
          v16h a2 = *(const v16h*)(ap + 2 * 54 * 512);
          v16h a3 = *(const v16h*)(ap + 3 * 54 * 512);
          acc0 = __builtin_amdgcn_wmma_f32_16x16x32_f16(false, a0, false, bfrag, (short)0, acc0, false, false);
          acc1 = __builtin_amdgcn_wmma_f32_16x16x32_f16(false, a1, false, bfrag, (short)0, acc1, false, false);
          acc2 = __builtin_amdgcn_wmma_f32_16x16x32_f16(false, a2, false, bfrag, (short)0, acc2, false, false);
          acc3 = __builtin_amdgcn_wmma_f32_16x16x32_f16(false, a3, false, bfrag, (short)0, acc3, false, false);
        }
      }
    }
  }

  // epilogue: BN scale/bias + leaky; C-frag: VGPR v -> co = mt*16 + hi*8 + v, z = zl
  size_t zbase = ((((size_t)b * 32 + xx) * 32 + yy) * 32 + zl) * 64;
  v8f accs[4] = {acc0, acc1, acc2, acc3};
  #pragma unroll
  for (int mt = 0; mt < 4; ++mt) {
    int cobase = mt * 16 + hi * 8;
    size_t obase = zbase + cobase;
    if (gout_h) {
      v8h oh;
      #pragma unroll
      for (int v = 0; v < 8; ++v) {
        float y = accs[mt][v] * par[sc_off + cobase + v] + par[bs_off + cobase + v];
        y = (y >= 0.f) ? y : LEAK * y;
        oh[v] = (_Float16)y;
      }
      *(v8h*)(gout_h + obase) = oh;
    } else {
      v8f of;
      #pragma unroll
      for (int v = 0; v < 8; ++v) {
        float y = accs[mt][v] * par[sc_off + cobase + v] + par[bs_off + cobase + v];
        of[v] = (y >= 0.f) ? y : LEAK * y;
      }
      *(v8f*)(gout_f + obase) = of;
    }
  }
}

// ---------------- features -> channels-last f16 ----------------
__global__ void feat_to_cl(const float* __restrict__ f, _Float16* __restrict__ fcl) {
  int t = blockIdx.x * blockDim.x + threadIdx.x;   // B*64*NPTS, t == feature linear idx
  int n = t & (NPTS - 1);
  int ci = (t >> 12) & 63;
  int b = t >> 18;
  fcl[((size_t)(b * NPTS + n)) * 64 + ci] = (_Float16)f[t];
}

// ---------------- point MLP 64x64 GEMM via WMMA, 4 co-tiles per wave ----------------
__global__ void pf_wmma(const _Float16* __restrict__ fcl, const _Float16* __restrict__ ppf,
                        const float* __restrict__ pfb, float* __restrict__ ycl) {
  int tid = threadIdx.x;
  int lane = tid & 31;
  int w = blockIdx.x * 8 + (tid >> 5);    // 2048 point tiles over B*NPTS
  int lm = lane & 15, hi = lane >> 4;
  int p0 = w * 16;

  v8f acc0 = {}, acc1 = {}, acc2 = {}, acc3 = {};
  #pragma unroll
  for (int s = 0; s < 2; ++s) {
    v16h bfrag = *(const v16h*)(fcl + (size_t)(p0 + lm) * 64 + s * 32 + hi * 16);
    const _Float16* ap = ppf + (size_t)s * 512 + lane * 16;
    v16h a0 = *(const v16h*)(ap);
    v16h a1 = *(const v16h*)(ap + 2 * 512);
    v16h a2 = *(const v16h*)(ap + 4 * 512);
    v16h a3 = *(const v16h*)(ap + 6 * 512);
    acc0 = __builtin_amdgcn_wmma_f32_16x16x32_f16(false, a0, false, bfrag, (short)0, acc0, false, false);
    acc1 = __builtin_amdgcn_wmma_f32_16x16x32_f16(false, a1, false, bfrag, (short)0, acc1, false, false);
    acc2 = __builtin_amdgcn_wmma_f32_16x16x32_f16(false, a2, false, bfrag, (short)0, acc2, false, false);
    acc3 = __builtin_amdgcn_wmma_f32_16x16x32_f16(false, a3, false, bfrag, (short)0, acc3, false, false);
  }
  v8f accs[4] = {acc0, acc1, acc2, acc3};
  size_t pbase = (size_t)(p0 + lm) * 64;
  #pragma unroll
  for (int mt = 0; mt < 4; ++mt) {
    int cobase = mt * 16 + hi * 8;
    v8f o;
    #pragma unroll
    for (int v = 0; v < 8; ++v) o[v] = accs[mt][v] + pfb[cobase + v];
    *(v8f*)(ycl + pbase + cobase) = o;
  }
}

// ---------------- GroupNorm stats over (8 ch, N) per (b, g) ----------------
__global__ void gn_stats(const float* __restrict__ ycl, float* par) {
  int bid = blockIdx.x;          // b*8 + g
  int b = bid >> 3, g = bid & 7;
  __shared__ float rs[256], rq[256];
  float s = 0.f, q = 0.f;
  for (int i = threadIdx.x; i < 8 * NPTS; i += 256) {
    int n = i >> 3, j = i & 7;
    float v = ycl[((size_t)(b * NPTS + n)) * 64 + g * 8 + j];
    s += v; q += v * v;
  }
  rs[threadIdx.x] = s; rq[threadIdx.x] = q; __syncthreads();
  for (int off = 128; off > 0; off >>= 1) {
    if (threadIdx.x < off) { rs[threadIdx.x] += rs[threadIdx.x + off]; rq[threadIdx.x] += rq[threadIdx.x + off]; }
    __syncthreads();
  }
  if (threadIdx.x == 0) {
    float mu = rs[0] / (float)(8 * NPTS);
    float var = rq[0] / (float)(8 * NPTS) - mu * mu;
    par[O_GNMU + bid] = mu;
    par[O_GNINV + bid] = rsqrtf(var + EPS_);
  }
}

// ---------------- devox (trilinear) + GN-affine + Swish + fuse ----------------
__global__ void fuse_final(const float* __restrict__ grid3, const float* __restrict__ ycl,
                           const float* __restrict__ nc, const float* __restrict__ par,
                           const float* __restrict__ gng, const float* __restrict__ gnb,
                           float* __restrict__ out) {
  int t = blockIdx.x * blockDim.x + threadIdx.x;  // B*NPTS
  int b = t >> 12, n = t & (NPTS - 1);
  float d[3]; int i0[3], i1[3];
  for (int a = 0; a < 3; ++a) {
    float cp = nc[(b * 3 + a) * NPTS + n];
    float fl = floorf(cp);
    d[a] = cp - fl;
    i0[a] = (int)fl;
    i1[a] = min(i0[a] + 1, RV - 1);
  }
  float a64[64];
  #pragma unroll
  for (int c = 0; c < 64; ++c) a64[c] = 0.f;
  for (int corner = 0; corner < 8; ++corner) {
    int ax = (corner >> 2) & 1, ay = (corner >> 1) & 1, az = corner & 1;
    float wt = (ax ? d[0] : 1.f - d[0]) * (ay ? d[1] : 1.f - d[1]) * (az ? d[2] : 1.f - d[2]);
    int ix = ax ? i1[0] : i0[0], iy = ay ? i1[1] : i0[1], iz = az ? i1[2] : i0[2];
    const float* gp = grid3 + ((size_t)b * R3 + ix * (RV * RV) + iy * RV + iz) * 64;
    #pragma unroll
    for (int c = 0; c < 64; ++c) a64[c] += wt * gp[c];
  }
  const float* yp = ycl + ((size_t)(b * NPTS + n)) * 64;
  #pragma unroll
  for (int c = 0; c < 64; ++c) {
    int g = c >> 3;
    float yn = (yp[c] - par[O_GNMU + b * 8 + g]) * par[O_GNINV + b * 8 + g] * gng[c] + gnb[c];
    float sw = yn / (1.f + __expf(-yn));
    out[((size_t)b * 64 + c) * NPTS + n] = a64[c] + sw;
  }
}

extern "C" void kernel_launch(void* const* d_in, const int* in_sizes, int n_in,
                              void* d_out, int out_size, void* d_ws, size_t ws_size,
                              hipStream_t stream) {
  const float* features = (const float*)d_in[0];
  const float* coords   = (const float*)d_in[1];
  const float* temb     = (const float*)d_in[2];
  const float* c1w = (const float*)d_in[3];
  const float* c1b = (const float*)d_in[4];
  const float* bn1g = (const float*)d_in[5];
  const float* bn1b = (const float*)d_in[6];
  const float* bn1m = (const float*)d_in[7];
  const float* bn1v = (const float*)d_in[8];
  const float* c2w = (const float*)d_in[9];
  const float* c2b = (const float*)d_in[10];
  const float* bn2g = (const float*)d_in[11];
  const float* bn2b = (const float*)d_in[12];
  const float* bn2m = (const float*)d_in[13];
  const float* bn2v = (const float*)d_in[14];
  const float* pfw = (const float*)d_in[15];
  const float* pfb = (const float*)d_in[16];
  const float* gng = (const float*)d_in[17];
  const float* gnb = (const float*)d_in[18];
  float* out = (float*)d_out;

  char* ws = (char*)d_ws;
  // workspace layout (bytes)
  const size_t o_par  = 0;                               // 4096
  const size_t o_nc   = 4096;                            // 393216
  const size_t o_cnt  = o_nc  + 393216;                  // 1048576
  const size_t o_sums = o_cnt + 1048576;                 // 67108864 (aliased: conv2 f32 out)
  const size_t o_g1   = o_sums + 67108864;               // 33554432 f16
  const size_t o_g2   = o_g1  + 33554432;                // 33554432 f16
  const size_t o_pw1  = o_g2  + 33554432;                // 221184
  const size_t o_pw2  = o_pw1 + 221184;                  // 221184
  const size_t o_ppf  = o_pw2 + 221184;                  // 8192
  const size_t o_fcl  = o_ppf + 8192;                    // 4194304 f16
  const size_t o_ycl  = o_fcl + 4194304;                 // 8388608 f32

  float*    par  = (float*)(ws + o_par);
  float*    ncrd = (float*)(ws + o_nc);
  float*    cnt  = (float*)(ws + o_cnt);
  float*    sums = (float*)(ws + o_sums);   // also conv2 output grid (f32, channels-last)
  _Float16* g1   = (_Float16*)(ws + o_g1);
  _Float16* g2   = (_Float16*)(ws + o_g2);
  _Float16* pw1  = (_Float16*)(ws + o_pw1);
  _Float16* pw2  = (_Float16*)(ws + o_pw2);
  _Float16* ppf  = (_Float16*)(ws + o_ppf);
  _Float16* fcl  = (_Float16*)(ws + o_fcl);
  float*    ycl  = (float*)(ws + o_ycl);

  // zero scatter buffers (cnt + sums are contiguous)
  hipMemsetAsync(ws + o_cnt, 0, 1048576 + 67108864, stream);

  prep_params<<<1, 64, 0, stream>>>(c1b, bn1g, bn1b, bn1m, bn1v,
                                    c2b, bn2g, bn2b, bn2m, bn2v, par);
  pack_weights<<<880, 256, 0, stream>>>(c1w, c2w, pfw, pw1, pw2, ppf);
  coord_stats<<<BN_, 256, 0, stream>>>(coords, par);
  voxelize<<<(BN_ * NPTS) / 256, 256, 0, stream>>>(coords, features, par, ncrd, cnt, sums);
  finalize_grid<<<(BN_ * R3) / 256, 256, 0, stream>>>(sums, cnt, g1);

  // conv1: g1(f16) -> g2(f16), BN1+leaky fused   (16384 waves / 8 per block)
  conv3d_wmma<<<2048, 256, 0, stream>>>(g1, pw1, par, O_BN1SC, O_BN1BS, g2, nullptr);
  // conv2: g2(f16) -> sums region (f32), BN2+leaky fused
  conv3d_wmma<<<2048, 256, 0, stream>>>(g2, pw2, par, O_BN2SC, O_BN2BS, nullptr, sums);

  feat_to_cl<<<(BN_ * 64 * NPTS) / 256, 256, 0, stream>>>(features, fcl);
  pf_wmma<<<256, 256, 0, stream>>>(fcl, ppf, pfb, ycl);
  gn_stats<<<64, 256, 0, stream>>>(ycl, par);
  fuse_final<<<(BN_ * NPTS) / 256, 256, 0, stream>>>(sums, ycl, ncrd, par, gng, gnb, out);

  // passthrough outputs: coords then temb after fused [8,64,4096]
  hipMemcpyAsync(out + 2097152, coords, 98304 * sizeof(float),
                 hipMemcpyDeviceToDevice, stream);
  hipMemcpyAsync(out + 2195456, temb, 512 * sizeof(float),
                 hipMemcpyDeviceToDevice, stream);
}
